// DeepKernelAttention_14130442404179
// MI455X (gfx1250) — matrix operand order repaired and checked by reading, not compiled, not run on hardware
//
#include <hip/hip_runtime.h>
#include <hip/hip_bf16.h>
#include <math.h>

// ---------------------------------------------------------------------------
// DeepKernelAttention for MI455X (gfx1250, wave32, WMMA).
//   B=16, Q=K=8192, D_QK=D_V=64, H=8, DP=128, VP=64, dh=16, eh=8
// Pipeline:
//   prep   : f32 weights -> f16, transposed [n][k] (WMMA B-operand friendly)
//   keys_kv: per 64-key tile: k-MLP (WMMA) + mask, v-MLP (WMMA) + 1/len,
//            partial kv[h][16][8] per tile -> workspace
//   kv_red : deterministic reduction over 128 key tiles -> kv[b][h][16][8]
//   query  : q-MLP (WMMA) + ctx = q*kv + LayerNorm -> out
// ---------------------------------------------------------------------------

typedef __attribute__((ext_vector_type(16))) _Float16 v16h;
typedef __attribute__((ext_vector_type(8)))  float    v8f;
typedef __attribute__((ext_vector_type(4)))  _Float16 v4h;

#define BATCH 16
#define SEQQ 8192
#define SEQK 8192
#define DQK 64
#define DV 64
#define NH 8
#define DP 128
#define VP 64
#define ROWS 64
#define KTILES (SEQK / ROWS)   // 128
#define LN_EPS 1e-6f
#define INV_SCALE 0.3535533905932738f   // 1 / 64^0.25

// Branch-free tanh: prefer the gfx1250 hardware v_tanh_f32; otherwise a
// saturating exp2/rcp formulation (no divergent saveexec blocks either way).
__device__ __forceinline__ float fast_tanh(float y) {
#if __has_builtin(__builtin_amdgcn_tanhf)
  return __builtin_amdgcn_tanhf(y);
#else
  float ay = fabsf(y);
  // tanh(|y|) = 1 - 2 / (exp(2|y|) + 1);  exp(2a) = exp2(a * 2*log2(e))
#if __has_builtin(__builtin_amdgcn_exp2f)
  float e = __builtin_amdgcn_exp2f(ay * 2.8853900817779268f);
#else
  float e = exp2f(ay * 2.8853900817779268f);
#endif
#if __has_builtin(__builtin_amdgcn_rcpf)
  float r = __builtin_amdgcn_rcpf(e + 1.0f);
#else
  float r = 1.0f / (e + 1.0f);
#endif
  float t = 1.0f - 2.0f * r;            // e==inf -> r==0 -> t==1
  return copysignf(t, y);
#endif
}

__device__ __forceinline__ float gelu_tanh(float x) {
  float x3 = x * x * x;
  float t = fast_tanh(0.7978845608028654f * (x + 0.044715f * x3));
  return 0.5f * x * (1.0f + t);
}

// One 16x16 output tile of X(MxK,f16 LDS) @ Wt(Nxldk,f16 global, [n][k]),
// accumulated f32 through v_wmma_f32_16x16x32_f16.
// Fragment layouts per CDNA5 ISA 7.12.2 (16-bit A 16x32; B column-striped).
__device__ __forceinline__ v8f wmma_tile_f16(const _Float16* X, int ldx,
                                             const _Float16* Wt, int ldk,
                                             int mt, int nt, int kdim, int lane) {
  v8f acc = {};
  const int mrow = mt * 16 + (lane & 15);
  const int khiA = (lane & 16) ? 8 : 0;
  const int ncol = nt * 16 + (lane & 15);
  const int khiB = (lane & 16) ? 8 : 0;
  for (int k0 = 0; k0 < kdim; k0 += 32) {
    v16h a, b;
#pragma unroll
    for (int i = 0; i < 8; ++i) {
      a[i]     = X[mrow * ldx + k0 + khiA + i];
      a[i + 8] = X[mrow * ldx + k0 + 16 + khiA + i];
    }
#pragma unroll
    for (int i = 0; i < 8; ++i) {
      b[i]     = Wt[ncol * ldk + k0 + khiB + i];
      b[i + 8] = Wt[ncol * ldk + k0 + 16 + khiB + i];
    }
    acc = __builtin_amdgcn_wmma_f32_16x16x32_f16(
        /*neg_a=*/false, a, /*neg_b=*/false, b,
        /*c_mod=*/(short)0, acc, /*reuse_a=*/false, /*reuse_b=*/false);
  }
  return acc;
}

__device__ __forceinline__ v4h cvt4(float4 v) {
  v4h h;
  h[0] = (_Float16)v.x; h[1] = (_Float16)v.y;
  h[2] = (_Float16)v.z; h[3] = (_Float16)v.w;
  return h;
}

// ---------------------------------------------------------------------------
// prep: transpose+convert weights to f16 [n][k]
// ---------------------------------------------------------------------------
__global__ void dka_prep(const float* __restrict__ Wq1, const float* __restrict__ Wq2,
                         const float* __restrict__ Wv1, const float* __restrict__ Wv2,
                         _Float16* __restrict__ Wtq1, _Float16* __restrict__ Wtq2,
                         _Float16* __restrict__ Wtv1, _Float16* __restrict__ Wtv2) {
  int gid = blockIdx.x * blockDim.x + threadIdx.x;
  if (gid < 16384) {                       // Wq1: (128,128) [k][n] -> [n][k]
    int n = gid >> 7, k = gid & 127;
    Wtq1[gid] = (_Float16)Wq1[k * 128 + n];
  } else if (gid < 32768) {                // Wq2: (128,128)
    int g = gid - 16384, n = g >> 7, k = g & 127;
    Wtq2[g] = (_Float16)Wq2[k * 128 + n];
  } else if (gid < 40960) {                // Wv1: (64,128) -> [n<128][k<64]
    int g = gid - 32768, n = g >> 6, k = g & 63;
    Wtv1[g] = (_Float16)Wv1[k * 128 + n];
  } else if (gid < 49152) {                // Wv2: (128,64) -> [n<64][k<128]
    int g = gid - 40960, n = g >> 7, k = g & 127;
    Wtv2[g] = (_Float16)Wv2[k * 64 + n];
  }
}

// ---------------------------------------------------------------------------
// keys_kv: k-MLP + v-MLP on a 64-row key tile, emit partial kv summary
// grid = (KTILES, BATCH), 256 threads (8 waves)
// ---------------------------------------------------------------------------
__global__ __launch_bounds__(256)
void dka_keys_kv(const float* __restrict__ ks, const float* __restrict__ ks_s,
                 const float* __restrict__ vs, const int* __restrict__ valid_lens,
                 const _Float16* __restrict__ Wtq1, const float* __restrict__ bq1,
                 const _Float16* __restrict__ Wtq2, const float* __restrict__ bq2,
                 const _Float16* __restrict__ Wtv1, const float* __restrict__ bv1,
                 const _Float16* __restrict__ Wtv2, const float* __restrict__ bv2,
                 float* __restrict__ kvpart) {
  __shared__ _Float16 Xl[ROWS][DP];   // concat(ks_s,ks), later Kf
  __shared__ _Float16 H1[ROWS][DP];   // hidden activations (both MLPs)
  __shared__ _Float16 Vx[ROWS][DV];   // vs, later Vf
  const int tid = threadIdx.x;
  const int lane = tid & 31, wave = tid >> 5;
  const int b = blockIdx.y;
  const int rowbase = blockIdx.x * ROWS;
  const int vl = valid_lens[b];
  const float inv_vl = 1.0f / (float)vl;

  // stage inputs -> LDS (f16), float4 global loads / b64 LDS stores
#pragma unroll
  for (int o = tid; o < ROWS * DP / 4; o += 256) {   // 2048 groups, 8 iters
    int r = o >> 5, c4 = (o & 31) * 4;
    size_t gi = ((size_t)b * SEQK + rowbase + r) * DQK;
    float4 v = (c4 < DQK) ? *(const float4*)(ks_s + gi + c4)
                          : *(const float4*)(ks + gi + (c4 - DQK));
    *(v4h*)&Xl[r][c4] = cvt4(v);
  }
#pragma unroll
  for (int o = tid; o < ROWS * DV / 4; o += 256) {   // 1024 groups, 4 iters
    int r = o >> 4, c4 = (o & 15) * 4;
    float4 v = *(const float4*)(vs + ((size_t)b * SEQK + rowbase + r) * DV + c4);
    *(v4h*)&Vx[r][c4] = cvt4(v);
  }
  __syncthreads();

  // k layer 1: H1 = gelu(X @ Wq1 + bq1)        (64x128, K=128)
  for (int t = wave; t < 32; t += 8) {
    int mt = t >> 3, nt = t & 7;
    v8f acc = wmma_tile_f16(&Xl[0][0], DP, Wtq1, DP, mt, nt, DP, lane);
    int ncol = nt * 16 + (lane & 15);
    int mhi = (lane & 16) ? 8 : 0;
    float bias = bq1[ncol];
#pragma unroll
    for (int j = 0; j < 8; ++j)
      H1[mt * 16 + mhi + j][ncol] = (_Float16)gelu_tanh(acc[j] + bias);
  }
  __syncthreads();

  // k layer 2: Kf = mask * (H1 @ Wq2 + bq2) / scale   -> reuse Xl
  for (int t = wave; t < 32; t += 8) {
    int mt = t >> 3, nt = t & 7;
    v8f acc = wmma_tile_f16(&H1[0][0], DP, Wtq2, DP, mt, nt, DP, lane);
    int ncol = nt * 16 + (lane & 15);
    int mhi = (lane & 16) ? 8 : 0;
    float bias = bq2[ncol];
#pragma unroll
    for (int j = 0; j < 8; ++j) {
      int m = mt * 16 + mhi + j;
      float val = ((rowbase + m) < vl) ? (acc[j] + bias) * INV_SCALE : 0.0f;
      Xl[m][ncol] = (_Float16)val;
    }
  }
  __syncthreads();

  // v layer 1: H1 = gelu(Vx @ Wv1 + bv1)       (64x128, K=64)
  for (int t = wave; t < 32; t += 8) {
    int mt = t >> 3, nt = t & 7;
    v8f acc = wmma_tile_f16(&Vx[0][0], DV, Wtv1, DV, mt, nt, DV, lane);
    int ncol = nt * 16 + (lane & 15);
    int mhi = (lane & 16) ? 8 : 0;
    float bias = bv1[ncol];
#pragma unroll
    for (int j = 0; j < 8; ++j)
      H1[mt * 16 + mhi + j][ncol] = (_Float16)gelu_tanh(acc[j] + bias);
  }
  __syncthreads();

  // v layer 2: Vf = (H1 @ Wv2 + bv2) / vl      (64x64, K=128) -> reuse Vx
  for (int t = wave; t < 16; t += 8) {
    int mt = t >> 2, nt = t & 3;
    v8f acc = wmma_tile_f16(&H1[0][0], DP, Wtv2, DP, mt, nt, DP, lane);
    int ncol = nt * 16 + (lane & 15);
    int mhi = (lane & 16) ? 8 : 0;
    float bias = bv2[ncol];
#pragma unroll
    for (int j = 0; j < 8; ++j)
      Vx[mt * 16 + mhi + j][ncol] = (_Float16)((acc[j] + bias) * inv_vl);
  }
  __syncthreads();

  // partial kv[h][d][e] = sum_r Kf[r][h*16+d] * Vf[r][h*8+e]  (1024 outputs)
  float* mypart = kvpart + ((size_t)b * KTILES + blockIdx.x) * (NH * 16 * 8);
#pragma unroll
  for (int j = 0; j < 4; ++j) {
    int o = tid + 256 * j;
    int h = o >> 7, d = (o >> 3) & 15, e = o & 7;
    float s = 0.0f;
    for (int r = 0; r < ROWS; ++r)
      s += (float)Xl[r][h * 16 + d] * (float)Vx[r][h * 8 + e];
    mypart[o] = s;
  }
}

// ---------------------------------------------------------------------------
// kv reduce: sum partials over KTILES (deterministic order)
// grid = 64 blocks x 256 threads -> 16384 = B*H*16*8 outputs
// ---------------------------------------------------------------------------
__global__ void dka_kv_reduce(const float* __restrict__ kvpart, float* __restrict__ kv) {
  int gid = blockIdx.x * 256 + threadIdx.x;   // 0..16383
  int b = gid >> 10, o = gid & 1023;
  const float* p = kvpart + ((size_t)b * KTILES) * 1024 + o;
  float s = 0.0f;
  for (int t = 0; t < KTILES; ++t) s += p[(size_t)t * 1024];
  kv[gid] = s;
}

// ---------------------------------------------------------------------------
// query: q-MLP + ctx = q*kv + LayerNorm
// grid = (SEQQ/ROWS, BATCH), 256 threads (8 waves)
// ---------------------------------------------------------------------------
__global__ __launch_bounds__(256)
void dka_query_ctx(const float* __restrict__ qs, const float* __restrict__ qs_s,
                   const _Float16* __restrict__ Wtq1, const float* __restrict__ bq1,
                   const _Float16* __restrict__ Wtq2, const float* __restrict__ bq2,
                   const float* __restrict__ kv,
                   const float* __restrict__ ln_scale, const float* __restrict__ ln_bias,
                   float* __restrict__ out) {
  __shared__ _Float16 Xl[ROWS][DP];   // concat(qs_s,qs), later Qf
  __shared__ _Float16 H1[ROWS][DP];
  __shared__ float kvl[NH * 16 * 8];  // 4 KB
  __shared__ float ctx[ROWS][VP];
  __shared__ float muA[ROWS], rsA[ROWS];
  const int tid = threadIdx.x;
  const int lane = tid & 31, wave = tid >> 5;
  const int b = blockIdx.y;
  const int rowbase = blockIdx.x * ROWS;

#pragma unroll
  for (int o = tid; o < ROWS * DP / 4; o += 256) {
    int r = o >> 5, c4 = (o & 31) * 4;
    size_t gi = ((size_t)b * SEQQ + rowbase + r) * DQK;
    float4 v = (c4 < DQK) ? *(const float4*)(qs_s + gi + c4)
                          : *(const float4*)(qs + gi + (c4 - DQK));
    *(v4h*)&Xl[r][c4] = cvt4(v);
  }
  for (int o = tid; o < NH * 16 * 8; o += 256)
    kvl[o] = kv[(size_t)b * 1024 + o];
  __syncthreads();

  // q layer 1
  for (int t = wave; t < 32; t += 8) {
    int mt = t >> 3, nt = t & 7;
    v8f acc = wmma_tile_f16(&Xl[0][0], DP, Wtq1, DP, mt, nt, DP, lane);
    int ncol = nt * 16 + (lane & 15);
    int mhi = (lane & 16) ? 8 : 0;
    float bias = bq1[ncol];
#pragma unroll
    for (int j = 0; j < 8; ++j)
      H1[mt * 16 + mhi + j][ncol] = (_Float16)gelu_tanh(acc[j] + bias);
  }
  __syncthreads();

  // q layer 2: Qf = (H1 @ Wq2 + bq2) / scale -> reuse Xl
  for (int t = wave; t < 32; t += 8) {
    int mt = t >> 3, nt = t & 7;
    v8f acc = wmma_tile_f16(&H1[0][0], DP, Wtq2, DP, mt, nt, DP, lane);
    int ncol = nt * 16 + (lane & 15);
    int mhi = (lane & 16) ? 8 : 0;
    float bias = bq2[ncol];
#pragma unroll
    for (int j = 0; j < 8; ++j)
      Xl[mt * 16 + mhi + j][ncol] = (_Float16)((acc[j] + bias) * INV_SCALE);
  }
  __syncthreads();

  // ctx[r][h*8+e] = sum_d Qf[r][h*16+d] * kv[h][d][e]
#pragma unroll
  for (int j = 0; j < 16; ++j) {
    int o = tid + 256 * j;
    int r = o >> 6, c = o & 63, h = c >> 3, e = c & 7;
    float s = 0.0f;
#pragma unroll
    for (int d = 0; d < 16; ++d)
      s += (float)Xl[r][h * 16 + d] * kvl[(h * 16 + d) * 8 + e];
    ctx[r][c] = s;
  }
  __syncthreads();

  // LayerNorm stats per row
  if (tid < ROWS) {
    float m = 0.0f;
    for (int c = 0; c < VP; ++c) m += ctx[tid][c];
    m *= (1.0f / VP);
    float v = 0.0f;
    for (int c = 0; c < VP; ++c) { float d = ctx[tid][c] - m; v += d * d; }
    v *= (1.0f / VP);
    muA[tid] = m;
    rsA[tid] = rsqrtf(v + LN_EPS);
  }
  __syncthreads();

  // normalize + affine, coalesced float4 stores
#pragma unroll
  for (int j = 0; j < 4; ++j) {
    int o = tid + 256 * j;               // 1024 float4 groups
    int r = o >> 4, c4 = (o & 15) * 4;
    float m = muA[r], rs = rsA[r];
    float4 res;
    res.x = (ctx[r][c4 + 0] - m) * rs * ln_scale[c4 + 0] + ln_bias[c4 + 0];
    res.y = (ctx[r][c4 + 1] - m) * rs * ln_scale[c4 + 1] + ln_bias[c4 + 1];
    res.z = (ctx[r][c4 + 2] - m) * rs * ln_scale[c4 + 2] + ln_bias[c4 + 2];
    res.w = (ctx[r][c4 + 3] - m) * rs * ln_scale[c4 + 3] + ln_bias[c4 + 3];
    *(float4*)(out + ((size_t)b * SEQQ + rowbase + r) * VP + c4) = res;
  }
}

// ---------------------------------------------------------------------------
extern "C" void kernel_launch(void* const* d_in, const int* in_sizes, int n_in,
                              void* d_out, int out_size, void* d_ws, size_t ws_size,
                              hipStream_t stream) {
  (void)in_sizes; (void)n_in; (void)out_size; (void)ws_size;
  const float* qs        = (const float*)d_in[0];
  const float* ks        = (const float*)d_in[1];
  const float* vs        = (const float*)d_in[2];
  const float* qs_s      = (const float*)d_in[3];
  const float* ks_s      = (const float*)d_in[4];
  const int*   valid_lens= (const int*)d_in[5];
  const float* Wq1       = (const float*)d_in[6];
  const float* bq1       = (const float*)d_in[7];
  const float* Wq2       = (const float*)d_in[8];
  const float* bq2       = (const float*)d_in[9];
  const float* Wv1       = (const float*)d_in[10];
  const float* bv1       = (const float*)d_in[11];
  const float* Wv2       = (const float*)d_in[12];
  const float* bv2       = (const float*)d_in[13];
  const float* ln_scale  = (const float*)d_in[14];
  const float* ln_bias   = (const float*)d_in[15];
  float* out = (float*)d_out;

  // workspace layout (bytes):
  //   [0,32768)       Wtq1 f16
  //   [32768,65536)   Wtq2 f16
  //   [65536,81920)   Wtv1 f16
  //   [81920,98304)   Wtv2 f16
  //   [98304,163840)  kv f32 (B*H*16*8)
  //   [163840,+8MB)   kvpart f32 (B*KTILES*1024)
  char* ws = (char*)d_ws;
  _Float16* Wtq1 = (_Float16*)(ws + 0);
  _Float16* Wtq2 = (_Float16*)(ws + 32768);
  _Float16* Wtv1 = (_Float16*)(ws + 65536);
  _Float16* Wtv2 = (_Float16*)(ws + 81920);
  float* kvbuf   = (float*)(ws + 98304);
  float* kvpart  = (float*)(ws + 163840);

  dka_prep<<<192, 256, 0, stream>>>(Wq1, Wq2, Wv1, Wv2, Wtq1, Wtq2, Wtv1, Wtv2);

  dim3 gridA(KTILES, BATCH);
  dka_keys_kv<<<gridA, 256, 0, stream>>>(ks, ks_s, vs, valid_lens,
                                         Wtq1, bq1, Wtq2, bq2,
                                         Wtv1, bv1, Wtv2, bv2, kvpart);

  dka_kv_reduce<<<64, 256, 0, stream>>>(kvpart, kvbuf);

  dim3 gridB(SEQQ / ROWS, BATCH);
  dka_query_ctx<<<gridB, 256, 0, stream>>>(qs, qs_s, Wtq1, bq1, Wtq2, bq2,
                                           kvbuf, ln_scale, ln_bias, out);
}
